// BetweennessModule_2894807957731
// MI455X (gfx1250) — matrix operand order repaired and checked by reading, not compiled, not run on hardware
//
#include <hip/hip_runtime.h>

// ---------------------------------------------------------------------------
// Problem dims (fixed by the reference): B=8, S=4096, D=512
// ---------------------------------------------------------------------------
#define BATCH 8
#define SEQ   4096
#define DIM   512
#define MROWS 64      // output rows (M) per workgroup in the GEMM stage
#define NOFF  20      // dot-product offsets 1..20 (covers 1..10 and 2*off)

typedef __attribute__((ext_vector_type(16))) __bf16 v16bf;
typedef __attribute__((ext_vector_type(8)))  float  v8f;

union V16BF {
    v16bf v;
    uint4 q[2];
    unsigned int u[8];
};

// round-to-nearest-even f32 -> bf16, packed pair (lo in [15:0], hi in [31:16])
__device__ __forceinline__ unsigned int pack2_bf16(float lo, float hi) {
    unsigned int a = __float_as_uint(lo);
    unsigned int b = __float_as_uint(hi);
    a = (a + 0x7FFFu + ((a >> 16) & 1u)) >> 16;
    b = (b + 0x7FFFu + ((b >> 16) & 1u)) >> 16;
    return (b << 16) | (a & 0xFFFFu);
}

// ---------------------------------------------------------------------------
// Stage 0: one-shot f32 -> bf16 conversion of x and W into workspace.
// Memory-bound; the VALU cost of RNE packing hides behind HBM streaming.
// Each thread converts 8 elements (2x float4 in, 1x uint4 out).
// ---------------------------------------------------------------------------
__global__ __launch_bounds__(256) void convert_bf16_kernel(
    const float* __restrict__ x, const float* __restrict__ W,
    unsigned short* __restrict__ xbf, unsigned short* __restrict__ wbf)
{
    const size_t xElems = (size_t)BATCH * SEQ * DIM;     // 16,777,216
    const size_t gid    = (size_t)blockIdx.x * 256 + threadIdx.x;
    const size_t base   = gid * 8;

    const float* src;
    unsigned short* dst;
    size_t e;
    if (base < xElems) { src = x; dst = xbf; e = base; }
    else               { src = W; dst = wbf; e = base - xElems; }

    const float4 va = *(const float4*)(src + e);
    const float4 vb = *(const float4*)(src + e + 4);
    uint4 o;
    o.x = pack2_bf16(va.x, va.y);
    o.y = pack2_bf16(va.z, va.w);
    o.z = pack2_bf16(vb.x, vb.y);
    o.w = pack2_bf16(vb.z, vb.w);
    *(uint4*)(dst + e) = o;
}

// ---------------------------------------------------------------------------
// Stage 1: h = x @ W^T + b  (bf16 WMMA, f32 accum), fused LayerNorm + L2-norm.
// Block = 256 threads = 8 waves arranged 2(M) x 4(N); block does 64 x 512.
// Wave computes a 32(M) x 128(N) macro-tile = 16 WMMA C-tiles:
//   per K-step (K=32): 2 A fragments (rows 0-15 / 16-31 of the wave tile),
//   8 B fragments, each B feeding TWO wmma ops -> 20 b128 loads : 16 wmma.
// Fragments load directly in ISA register layout (2x b128 each):
//   A (16x32 bf16): lane holds K = {8*halfk..+8} and {16+8*halfk..+8}
//   B (32x16 bf16): lane holds K = {8*halfk..+8} and {16+8*halfk..+8}  (W row n = B column n)
// ---------------------------------------------------------------------------
__global__ __launch_bounds__(256) void gemm_ln_kernel(
    const unsigned short* __restrict__ xbf, const unsigned short* __restrict__ wbf,
    const float* __restrict__ bias, const float* __restrict__ gamma,
    const float* __restrict__ beta, float* __restrict__ cn)
{
    __shared__ float hrow[MROWS][DIM + 16];   // +16 pad vs bank conflicts

    const int tid   = threadIdx.x;
    const int wave  = tid >> 5;       // 0..7
    const int waveM = wave >> 2;      // 0..1  (M half: 32 rows each)
    const int waveN = wave & 3;       // 0..3  (N quarter: 128 cols each)
    const int lane  = tid & 31;
    const int halfk = lane >> 4;      // which K-half this lane holds
    const int lrow  = lane & 15;      // M (for A) / N (for B,C) within tile

    const long rowBase = (long)blockIdx.x * MROWS;   // global M base (= b*S+s)
    const int  nbase   = waveN * 128;

    v8f acc[16] = {};

    const unsigned short* aRow = xbf + (rowBase + waveM * 32 + lrow) * DIM;
    const unsigned short* wRow = wbf + (size_t)(nbase + lrow) * DIM;
    const int aOff = halfk * 8;       // element offset within K-block
    const int bOff = halfk * 8;

    for (int ks = 0; ks < DIM / 32; ++ks) {
        const int k0 = ks * 32;
        V16BF A0, A1;                 // two 16-row A fragments of the 32-row tile
        A0.q[0] = *(const uint4*)(aRow + k0 + aOff);
        A0.q[1] = *(const uint4*)(aRow + k0 + aOff + 16);
        A1.q[0] = *(const uint4*)(aRow + 16 * DIM + k0 + aOff);
        A1.q[1] = *(const uint4*)(aRow + 16 * DIM + k0 + aOff + 16);
#pragma unroll
        for (int t = 0; t < 8; ++t) {
            V16BF Bm;                 // t*16*DIM folds into the 24-bit ioffset
            Bm.q[0] = *(const uint4*)(wRow + t * 16 * DIM + k0 + bOff);
            Bm.q[1] = *(const uint4*)(wRow + t * 16 * DIM + k0 + bOff + 16);
            acc[t] = __builtin_amdgcn_wmma_f32_16x16x32_bf16(
                false, A0.v, false, Bm.v, (short)0, acc[t], false, false);
            acc[8 + t] = __builtin_amdgcn_wmma_f32_16x16x32_bf16(
                false, A1.v, false, Bm.v, (short)0, acc[8 + t], false, false);
        }
    }

    // --- spill C tiles (+bias) to LDS. C layout: VGPR r, lane l ->
    //     M = r + 8*(l>>4), N = l&15.
#pragma unroll
    for (int h = 0; h < 2; ++h) {
#pragma unroll
        for (int t = 0; t < 8; ++t) {
            const int n  = nbase + t * 16 + lrow;
            const float bn = bias[n];
#pragma unroll
            for (int r = 0; r < 8; ++r) {
                const int m = waveM * 32 + h * 16 + r + (halfk << 3);
                hrow[m][n] = acc[h * 8 + t][r] + bn;
            }
        }
    }
    __syncthreads();

    // --- fused LayerNorm + L2 normalize; wave w handles rows 8w..8w+7
#pragma unroll
    for (int rr = 0; rr < 8; ++rr) {
        const int m = wave * 8 + rr;
        float s = 0.f, sq = 0.f;
        for (int c = lane; c < DIM; c += 32) {
            const float v = hrow[m][c];
            s += v; sq = fmaf(v, v, sq);
        }
#pragma unroll
        for (int o = 16; o >= 1; o >>= 1) { s += __shfl_xor(s, o); sq += __shfl_xor(sq, o); }
        const float mean = s * (1.0f / DIM);
        const float var  = sq * (1.0f / DIM) - mean * mean;
        const float inv  = rsqrtf(var + 1e-5f);

        float n2 = 0.f;
        for (int c = lane; c < DIM; c += 32) {
            const float v = (hrow[m][c] - mean) * inv * gamma[c] + beta[c];
            hrow[m][c] = v;
            n2 = fmaf(v, v, n2);
        }
#pragma unroll
        for (int o = 16; o >= 1; o >>= 1) n2 += __shfl_xor(n2, o);
        const float rn = 1.0f / fmaxf(sqrtf(n2), 1e-12f);

        float* dst = cn + (rowBase + m) * DIM;
        for (int c = lane; c < DIM; c += 32) dst[c] = hrow[m][c] * rn;
    }
}

// ---------------------------------------------------------------------------
// Stage 2a: dots[b][off-1][t] = cn[b,t,:] . cn[b,t+off,:]   (off = 1..NOFF)
// One wave per dot product; cn (64 MB) lives entirely in the 192 MB L2.
// ---------------------------------------------------------------------------
__global__ __launch_bounds__(256) void dots_kernel(
    const float* __restrict__ cn, float* __restrict__ dots)
{
    const int  wave = threadIdx.x >> 5;
    const int  lane = threadIdx.x & 31;
    const long gid  = (long)blockIdx.x * 8 + wave;       // linear dot index
    const int  t    = (int)(gid & (SEQ - 1));
    const int  off  = (int)((gid >> 12) % NOFF) + 1;     // SEQ = 2^12
    const int  b    = (int)(gid / ((long)SEQ * NOFF));

    float sum = 0.f;
    if (t + off < SEQ) {
        const float4* a = (const float4*)(cn + ((long)b * SEQ + t)       * DIM);
        const float4* c = (const float4*)(cn + ((long)b * SEQ + t + off) * DIM);
#pragma unroll
        for (int it = 0; it < 4; ++it) {
            const float4 va = a[it * 32 + lane];
            const float4 vc = c[it * 32 + lane];
            sum = fmaf(va.x, vc.x, sum);
            sum = fmaf(va.y, vc.y, sum);
            sum = fmaf(va.z, vc.z, sum);
            sum = fmaf(va.w, vc.w, sum);
        }
    }
#pragma unroll
    for (int o = 16; o >= 1; o >>= 1) sum += __shfl_xor(sum, o);
    if (lane == 0) dots[gid] = sum;
}

// ---------------------------------------------------------------------------
// Stage 2b: assemble betweenness scores, per-batch mean/std (ddof=1),
// gate & clip. One block per batch row; thread holds 16 positions.
// ---------------------------------------------------------------------------
__global__ __launch_bounds__(256) void bet_kernel(
    const float* __restrict__ dots, const float* __restrict__ gate,
    float* __restrict__ out)
{
    __shared__ float red[16];
    const int b   = blockIdx.x;
    const int tid = threadIdx.x;
    const float* dB = dots + (long)b * NOFF * SEQ;

    float vals[16];
    float s = 0.f, sq = 0.f;
#pragma unroll
    for (int c = 0; c < 16; ++c) {
        const int sp = tid + 256 * c;
        float acc = 0.f;
#pragma unroll
        for (int off = 1; off <= 10; ++off) {
            const int i = sp - off;
            if (i >= 0 && sp + off < SEQ) {   // i+2*off <= S-1
                const float dik = dB[(2 * off - 1) * SEQ + i];
                const float dij = dB[(off - 1) * SEQ + i];
                const float djk = dB[(off - 1) * SEQ + sp];
                const float direct = 1.f - dik;
                const float path   = 2.f - dij - djk;
                const float sd     = fmaxf(direct, 0.001f);
                acc += 1.f - (path - direct) / sd;
            }
        }
        acc *= 0.1f;                          // / WINDOW
        vals[c] = acc;
        s += acc; sq = fmaf(acc, acc, sq);
    }
#pragma unroll
    for (int o = 16; o >= 1; o >>= 1) { s += __shfl_xor(s, o); sq += __shfl_xor(sq, o); }
    const int wave = tid >> 5, lane = tid & 31;
    if (lane == 0) { red[wave] = s; red[8 + wave] = sq; }
    __syncthreads();
    if (tid == 0) {
        float ts = 0.f, tq = 0.f;
        for (int w = 0; w < 8; ++w) { ts += red[w]; tq += red[8 + w]; }
        red[0] = ts; red[1] = tq;
    }
    __syncthreads();
    const float mean = red[0] / SEQ;
    const float var  = fmaxf((red[1] - (float)SEQ * mean * mean) / (SEQ - 1), 0.f);
    const float g    = gate[0] /* * SCALE=1.0 */ / (sqrtf(var) + 1e-6f);
#pragma unroll
    for (int c = 0; c < 16; ++c) {
        const int sp = tid + 256 * c;
        float v = g * (vals[c] - mean);
        v = fminf(fmaxf(v, -2.f), 2.f);
        out[(long)b * SEQ + sp] = v;
    }
}

// ---------------------------------------------------------------------------
extern "C" void kernel_launch(void* const* d_in, const int* in_sizes, int n_in,
                              void* d_out, int out_size, void* d_ws, size_t ws_size,
                              hipStream_t stream)
{
    const float* x     = (const float*)d_in[0];   // (8,4096,512)
    const float* W     = (const float*)d_in[1];   // (512,512)
    const float* bias  = (const float*)d_in[2];   // (512,)
    const float* gamma = (const float*)d_in[3];   // (512,)
    const float* beta  = (const float*)d_in[4];   // (512,)
    const float* gate  = (const float*)d_in[5];   // (1,)
    float* out = (float*)d_out;                   // (8,4096)

    // workspace layout
    const size_t xElems = (size_t)BATCH * SEQ * DIM;          // 16.78M
    const size_t wElems = (size_t)DIM * DIM;                  // 262K
    char* ws = (char*)d_ws;
    float*          cnW   = (float*)ws;                                  // 64 MB
    float*          dotsW = (float*)(ws + xElems * sizeof(float));       // 2.62 MB
    unsigned short* xbf   = (unsigned short*)(ws + xElems * sizeof(float)
                              + (size_t)BATCH * NOFF * SEQ * sizeof(float)); // 32 MB
    unsigned short* wbf   = xbf + xElems;                                // 0.5 MB

    // Stage 0: convert x and W to bf16 (8 elements per thread)
    const size_t nConv = (xElems + wElems) / 8;               // 2,129,920
    convert_bf16_kernel<<<(unsigned)(nConv / 256), 256, 0, stream>>>(x, W, xbf, wbf);

    // Stage 1: 32768 rows / 64 rows-per-block
    gemm_ln_kernel<<<(BATCH * SEQ) / MROWS, 256, 0, stream>>>(xbf, wbf, bias,
                                                              gamma, beta, cnW);
    // Stage 2a: 8*20*4096 dots / 8 waves-per-block
    dots_kernel<<<(BATCH * NOFF * SEQ) / 8, 256, 0, stream>>>(cnW, dotsW);
    // Stage 2b
    bet_kernel<<<BATCH, 256, 0, stream>>>(dotsW, gate, out);
}
